// MultiHeadAttention_1589137900260
// MI455X (gfx1250) — matrix operand order repaired and checked by reading, not compiled, not run on hardware
//
#include <hip/hip_runtime.h>
#include <hip/hip_bf16.h>
#include <math.h>

// ---------------------------------------------------------------------------
// MHA with full-width per-head projections, bf16 WMMA pipeline for gfx1250.
// D = A x B + C via v_wmma_f32_16x16x32_bf16 (wave32, 16x16x32 tiles).
// Global->LDS staging uses CDNA5 GLOBAL_LOAD_ASYNC_TO_LDS_B128 + ASYNCcnt,
// double-buffered LDS tiles, 64x64 accumulator per wave (16 WMMA / 8 frags).
// ---------------------------------------------------------------------------

typedef __attribute__((ext_vector_type(16))) __bf16 v16bf;
typedef __attribute__((ext_vector_type(8)))  float  v8f;

#define DEVI __device__ __forceinline__

DEVI unsigned short f2bf(float f) {
  unsigned int u = __float_as_uint(f);
  u += 0x7FFFu + ((u >> 16) & 1u);          // round-to-nearest-even
  return (unsigned short)(u >> 16);
}
DEVI float bf2f(unsigned short h) {
  return __uint_as_float(((unsigned int)h) << 16);
}

// CDNA5 async copy: per-lane 16B global -> LDS, tracked by ASYNCcnt.
// LDS byte offset = low 32 bits of the generic pointer to __shared__.
DEVI void async_ld16(const unsigned short* g, unsigned ldsoff) {
  asm volatile("global_load_async_to_lds_b128 %0, %1, off"
               :: "v"(ldsoff), "v"(g) : "memory");
}
DEVI void wait_async0() {
#if __has_builtin(__builtin_amdgcn_s_wait_asynccnt)
  __builtin_amdgcn_s_wait_asynccnt(0);
#else
  asm volatile("s_wait_asynccnt 0x0" ::: "memory");
#endif
}
DEVI unsigned lds_off(const void* p) { return (unsigned)(size_t)p; }

// ------------------------------ fp32 -> bf16 -------------------------------
__global__ __launch_bounds__(256) void cvt_f32_bf16(
    const float* __restrict__ src, unsigned short* __restrict__ dst, long long n) {
  long long i = ((long long)blockIdx.x * 256 + threadIdx.x) * 4;
  if (i >= n) return;
  float4 v = *(const float4*)(src + i);
  unsigned short o[4] = {f2bf(v.x), f2bf(v.y), f2bf(v.z), f2bf(v.w)};
  *(uint2*)(dst + i) = *(const uint2*)o;
}

// ------------------------------ WMMA GEMM ----------------------------------
// C[M,N] = alpha * A[M,K] x B + bias, operands bf16 (C fp32 optional).
// B_IS_NK: B stored [N,K] (Linear weight layout, A x B^T). Else B is [K,N]
// (staged transposed into LDS).  CAUSAL: store -inf where col > row, and
// skip compute for fully masked tiles.  TRIMK: K-loop only up to row0+BM
// (P x V with causal probs).  Batch b: offset = (b%bmod)*s1 + (b/bmod)*s2.
template <bool B_IS_NK, bool CAUSAL, bool OUT_BF16, bool HAS_BIAS, bool TRIMK>
__global__ __launch_bounds__(256) void gemm_wmma(
    const unsigned short* __restrict__ A, const unsigned short* __restrict__ B,
    void* __restrict__ Cv, const float* __restrict__ bias,
    int lda, int ldb, int ldc, int M, int Nn, int K, int bmod,
    long long a1, long long a2, long long b1, long long b2,
    long long c1, long long c2, long long biasStride, float alpha) {
  constexpr int BM = 128, BN = 256, BK = 32, LDSTR = 40;  // 80B row stride
  __shared__ __align__(16) unsigned short As[2][BM * LDSTR];  // 2 x 10 KB
  __shared__ __align__(16) unsigned short Bs[2][BN * LDSTR];  // 2 x 20 KB

  const int bz = blockIdx.z;
  const int bi = bz % bmod, bj = bz / bmod;
  A += (long long)bi * a1 + (long long)bj * a2;
  B += (long long)bi * b1 + (long long)bj * b2;
  if (HAS_BIAS) bias += (long long)bj * biasStride;
  const long long cOff = (long long)bi * c1 + (long long)bj * c2;
  const int row0 = blockIdx.y * BM;
  const int col0 = blockIdx.x * BN;
  const int t = threadIdx.x;

  unsigned short* Cb = (unsigned short*)Cv;
  float*          Cf = (float*)Cv;

  if (CAUSAL && col0 > row0 + BM - 1) {  // tile entirely above diagonal
    const unsigned int pat = 0xFF80FF80u;  // two bf16 -inf
    uint4 q; q.x = pat; q.y = pat; q.z = pat; q.w = pat;
    const int r = t >> 1, cb = col0 + (t & 1) * 128;
    unsigned short* dst = Cb + cOff + (long long)(row0 + r) * ldc + cb;
#pragma unroll
    for (int i = 0; i < 16; i++) *(uint4*)(dst + i * 8) = q;
    return;
  }

  const int wave = t >> 5, lane = t & 31;
  const int wm = (wave & 1) * 64;    // 2x4 wave grid: 64x64 per wave
  const int wn = (wave >> 1) * 64;
  const int l15 = lane & 15;
  const int khalfA = (lane >> 4) * 8;   // A frag: K chunks {k0..+7, k0+16..+23}
  const int khalfB = (lane >> 4) * 8;   // B frag: K chunks {k0..+7, k0+16..+23}

  v8f zero = {};
  v8f acc[4][4];
#pragma unroll
  for (int i = 0; i < 4; i++)
#pragma unroll
    for (int j = 0; j < 4; j++) acc[i][j] = zero;

  const int ar = t >> 1;          // A staging: tile row 0..127
  const int ac = (t & 1) * 16;    // element offset within BK
  const int bkr = t >> 3;         // [K,N] staging: k row 0..31
  const int bnc = (t & 7) * 32;   // n offset 0..224

  const int Kend = TRIMK ? ((row0 + BM < K) ? (row0 + BM) : K) : K;

  auto stage = [&](int buf, int k0) {
    {  // A tile 128x32 -> LDS (async, no VGPR round trip)
      const unsigned short* ga = A + (long long)(row0 + ar) * lda + (k0 + ac);
      const unsigned d = lds_off(&As[buf][ar * LDSTR + ac]);
      async_ld16(ga, d);
      async_ld16(ga + 8, d + 16);
    }
    if (B_IS_NK) {  // B[N,K]: 256 rows, one row (32 K) per thread
      const unsigned short* gb = B + (long long)(col0 + t) * ldb + k0;
      const unsigned d = lds_off(&Bs[buf][t * LDSTR]);
#pragma unroll
      for (int c = 0; c < 4; c++) async_ld16(gb + 8 * c, d + 16 * c);
    } else {        // B[K,N]: coalesced read along n, transpose into LDS
      const unsigned short* gb = B + (long long)(k0 + bkr) * ldb + (col0 + bnc);
      uint4 v[4];
#pragma unroll
      for (int c = 0; c < 4; c++) v[c] = *(const uint4*)(gb + 8 * c);
      const unsigned short* e = (const unsigned short*)v;
#pragma unroll
      for (int c = 0; c < 32; c++) Bs[buf][(bnc + c) * LDSTR + bkr] = e[c];
    }
  };

  stage(0, 0);
  int buf = 0;
  for (int k0 = 0; k0 < Kend; k0 += BK) {
    wait_async0();       // staged tile (issued last iteration) has landed
    __syncthreads();     // visible to all waves (also covers transpose path)

    if (k0 + BK < Kend) {
      stage(buf ^ 1, k0 + BK);  // stream next tile while computing this one
      __builtin_prefetch(A + (long long)(row0 + ar) * lda + (k0 + 2 * BK + ac), 0, 1);
    }

    v16bf afr[4], bfr[4];
#pragma unroll
    for (int i = 0; i < 4; i++) {
      const unsigned short* p = &As[buf][(wm + i * 16 + l15) * LDSTR + khalfA];
      uint4* d = (uint4*)&afr[i];
      d[0] = *(const uint4*)(p);
      d[1] = *(const uint4*)(p + 16);
    }
#pragma unroll
    for (int j = 0; j < 4; j++) {
      const unsigned short* p = &Bs[buf][(wn + j * 16 + l15) * LDSTR + khalfB];
      uint4* d = (uint4*)&bfr[j];
      d[0] = *(const uint4*)(p);
      d[1] = *(const uint4*)(p + 16);
    }
#pragma unroll
    for (int i = 0; i < 4; i++)
#pragma unroll
      for (int j = 0; j < 4; j++)
        acc[i][j] = __builtin_amdgcn_wmma_f32_16x16x32_bf16(
            false, afr[i], false, bfr[j], (short)0, acc[i][j], false, false);
    buf ^= 1;
  }

  // Epilogue. C layout: VGPR e -> M = e (lanes 0-15) / e+8 (lanes 16-31), N = lane%16.
  const int rbase = row0 + wm + ((lane >= 16) ? 8 : 0);
  const int cbase = col0 + wn + l15;
#pragma unroll
  for (int i = 0; i < 4; i++) {
#pragma unroll
    for (int j = 0; j < 4; j++) {
      const int col = cbase + j * 16;
      const float bv = HAS_BIAS ? bias[col] : 0.0f;
#pragma unroll
      for (int e = 0; e < 8; e++) {
        const int row = rbase + i * 16 + e;
        float v = acc[i][j][e] * alpha + bv;
        if (CAUSAL && col > row) v = -INFINITY;
        if (OUT_BF16)
          Cb[cOff + (long long)row * ldc + col] = f2bf(v);
        else
          Cf[cOff + (long long)row * ldc + col] = v;
      }
    }
  }
}

// ------------------------------ row softmax --------------------------------
// One 256-thread block per row of 1024 bf16 scores; fp32 math; in place.
__global__ __launch_bounds__(256) void softmax_rows(
    unsigned short* __restrict__ P, int L) {
  __shared__ float red[256];
  unsigned short* p = P + (long long)blockIdx.x * L;
  const int t = threadIdx.x;
  float v[4], m = -INFINITY;
#pragma unroll
  for (int i = 0; i < 4; i++) {
    v[i] = bf2f(p[t + i * 256]);
    m = fmaxf(m, v[i]);
  }
  red[t] = m; __syncthreads();
  for (int s = 128; s > 0; s >>= 1) {
    if (t < s) red[t] = fmaxf(red[t], red[t + s]);
    __syncthreads();
  }
  m = red[0]; __syncthreads();
  float s = 0.f;
#pragma unroll
  for (int i = 0; i < 4; i++) { v[i] = __expf(v[i] - m); s += v[i]; }
  red[t] = s; __syncthreads();
  for (int s2 = 128; s2 > 0; s2 >>= 1) {
    if (t < s2) red[t] += red[t + s2];
    __syncthreads();
  }
  const float inv = 1.0f / red[0];
#pragma unroll
  for (int i = 0; i < 4; i++) p[t + i * 256] = f2bf(v[i] * inv);
}

// ------------------------------ launcher -----------------------------------
extern "C" void kernel_launch(void* const* d_in, const int* in_sizes, int n_in,
                              void* d_out, int out_size, void* d_ws, size_t ws_size,
                              hipStream_t stream) {
  const float* x  = (const float*)d_in[0];
  const float* Wq = (const float*)d_in[1];
  const float* bq = (const float*)d_in[2];
  const float* Wk = (const float*)d_in[3];
  const float* bk = (const float*)d_in[4];
  const float* Wv = (const float*)d_in[5];
  const float* bv = (const float*)d_in[6];
  const float* Wc = (const float*)d_in[7];
  const float* bc = (const float*)d_in[8];
  float* out = (float*)d_out;

  const int Nb = 8, S = 1024, E = 768, H = 12;
  const long long NS = (long long)Nb * S;   // 8192
  const long long EE = (long long)E * E;

  // Workspace carve-out (bf16 elements). ~878 MB total.
  unsigned short* ws = (unsigned short*)d_ws;
  long long o = 0;
  unsigned short* xb  = ws + o; o += NS * E;
  unsigned short* Wqb = ws + o; o += (long long)H * EE;
  unsigned short* Wkb = ws + o; o += (long long)H * EE;
  unsigned short* Wvb = ws + o; o += (long long)H * EE;
  unsigned short* Wcb = ws + o; o += (long long)H * EE;      // Wc is [E, H*E]
  unsigned short* Qb  = ws + o; o += (long long)H * NS * E;
  unsigned short* Kb  = ws + o; o += (long long)H * NS * E;
  unsigned short* Vb  = ws + o; o += (long long)H * NS * E;
  unsigned short* Pb  = ws + o; o += (long long)H * Nb * S * S;
  unsigned short* Cc  = ws + o; o += NS * (long long)H * E;  // concat [NS, H*E]

  auto cvt = [&](const float* s, unsigned short* d, long long n) {
    cvt_f32_bf16<<<dim3((unsigned)((n / 4 + 255) / 256)), 256, 0, stream>>>(s, d, n);
  };
  cvt(x,  xb,  NS * E);
  cvt(Wq, Wqb, (long long)H * EE);
  cvt(Wk, Wkb, (long long)H * EE);
  cvt(Wv, Wvb, (long long)H * EE);
  cvt(Wc, Wcb, (long long)H * EE);

  const dim3 blk(256);

  // --- QKV: per (h,n), [S,E] = xb[n] x W[h]^T + b[h]  (96 batched GEMMs) ---
  {
    dim3 g(E / 256, S / 128, H * Nb);
    const long long sSE = (long long)S * E, sNSE = (long long)Nb * S * E;
    gemm_wmma<true, false, true, true, false><<<g, blk, 0, stream>>>(
        xb, Wqb, Qb, bq, E, E, E, S, E, E, Nb,
        sSE, 0, 0, EE, sSE, sNSE, (long long)E, 1.0f);
    gemm_wmma<true, false, true, true, false><<<g, blk, 0, stream>>>(
        xb, Wkb, Kb, bk, E, E, E, S, E, E, Nb,
        sSE, 0, 0, EE, sSE, sNSE, (long long)E, 1.0f);
    gemm_wmma<true, false, true, true, false><<<g, blk, 0, stream>>>(
        xb, Wvb, Vb, bv, E, E, E, S, E, E, Nb,
        sSE, 0, 0, EE, sSE, sNSE, (long long)E, 1.0f);
  }

  // --- scores: P = causal( Q x K^T / sqrt(E) ) ---
  {
    dim3 g(S / 256, S / 128, H * Nb);
    const long long sSE = (long long)S * E, sNSE = (long long)Nb * S * E;
    const long long sSS = (long long)S * S, sNSS = (long long)Nb * S * S;
    gemm_wmma<true, true, true, false, false><<<g, blk, 0, stream>>>(
        Qb, Kb, Pb, nullptr, E, E, S, S, S, E, Nb,
        sSE, sNSE, sSE, sNSE, sSS, sNSS, 0, 1.0f / sqrtf((float)E));
  }

  // --- softmax over each of H*N*S rows of length S ---
  softmax_rows<<<dim3((unsigned)((long long)H * Nb * S)), 256, 0, stream>>>(Pb, S);

  // --- P x V -> concat[NS, H*E], K-loop trimmed by causality ---
  {
    dim3 g(E / 256, S / 128, H * Nb);
    const long long sSS = (long long)S * S, sNSS = (long long)Nb * S * S;
    const long long sSE = (long long)S * E, sNSE = (long long)Nb * S * E;
    gemm_wmma<false, false, true, false, true><<<g, blk, 0, stream>>>(
        Pb, Vb, Cc, nullptr, S, E, H * E, S, E, S, Nb,
        sSS, sNSS, sSE, sNSE, (long long)S * H * E, (long long)E, 0, 1.0f);
  }

  // --- out = concat x Wc^T + bc (fp32 output) ---
  {
    dim3 g(E / 256, (unsigned)(NS / 128), 1);
    gemm_wmma<true, false, false, true, false><<<g, blk, 0, stream>>>(
        Cc, Wcb, out, bc, H * E, H * E, E, (int)NS, E, H * E, 1,
        0, 0, 0, 0, 0, 0, 0, 1.0f);
  }
}
